// LeafBlockAttention_63376537420415
// MI455X (gfx1250) — matrix operand order repaired and checked by reading, not compiled, not run on hardware
//
#include <hip/hip_runtime.h>

typedef _Float16 v16h __attribute__((ext_vector_type(16)));
typedef _Float16 v8h  __attribute__((ext_vector_type(8)));
typedef _Float16 h4_t __attribute__((ext_vector_type(4)));
typedef float    v8f  __attribute__((ext_vector_type(8)));
typedef float    f4_t __attribute__((ext_vector_type(4)));

namespace {

constexpr int kNB = 512;
constexpr int kLQ = 64;
constexpr int kC  = 256;
constexpr int kH  = 8;
constexpr int kHD = 32;
constexpr float kScale = 0.17677669529663689f;  // 1/sqrt(32)

// ---- LDS layout, units of _Float16 ----
constexpr int QS = 776;   // qkv row stride (64 rows): Q [0,256) K [256,512) V [512,768)
constexpr int XS = 264;   // x-block / att-out row stride (64 rows)
constexpr int PS = 72;    // per-wave probs tile row stride (16 rows)
constexpr int OFF_QKV = 0;
constexpr int OFF_P   = OFF_QKV + 64 * QS;     // 49664
constexpr int OFF_XO  = OFF_P + kH * 16 * PS;  // 58880 (xb phase0/1 aliased w/ att-out 2/3/4)
constexpr int LDS_HALFS = OFF_XO + 64 * XS;    // 75776 halfs = 151552 bytes

__device__ __forceinline__ v8f zero8() {
  v8f z = {0.f, 0.f, 0.f, 0.f, 0.f, 0.f, 0.f, 0.f};
  return z;
}

__device__ __forceinline__ v16h cat16(v8h lo, v8h hh) {
  return __builtin_shufflevector(lo, hh, 0, 1, 2, 3, 4, 5, 6, 7, 8, 9, 10, 11, 12, 13, 14, 15);
}

// A fragment (16x32 f16) from row-major storage, stride in halfs (multiple of 8).
// Lane (r,hi): halves 0..15 = K[hi*16 .. hi*16+15] (contiguous).
__device__ __forceinline__ v16h load_a(const _Float16* base, int stride, int m0, int k0,
                                       int r, int hi) {
  const _Float16* p = base + (m0 + r) * stride + k0 + hi * 16;
  v8h lo = *reinterpret_cast<const v8h*>(p);
  v8h hh = *reinterpret_cast<const v8h*>(p + 8);
  return cat16(lo, hh);
}

// B fragment (32x16) from transpose-stored matrix: Bt is [N][K] row-major in LDS.
// Lane (r,hi): column n0+r, K = hi*16 + j (contiguous).
__device__ __forceinline__ v16h load_bt_lds(const _Float16* base, int stride, int n0, int k0,
                                            int r, int hi) {
  const _Float16* p = base + (n0 + r) * stride + k0 + hi * 16;
  v8h lo = *reinterpret_cast<const v8h*>(p);
  v8h hh = *reinterpret_cast<const v8h*>(p + 8);
  return cat16(lo, hh);
}

// B fragment (32x16) from fp32 weights stored [N][K] row-major in global (L2-resident);
// converted to f16 in-register.
__device__ __forceinline__ v16h load_bt_w(const float* W, int ldw, int n0, int k0,
                                          int r, int hi) {
  const float* p = W + (size_t)(n0 + r) * ldw + k0 + hi * 16;
  f4_t a = *reinterpret_cast<const f4_t*>(p);
  f4_t b = *reinterpret_cast<const f4_t*>(p + 4);
  f4_t c = *reinterpret_cast<const f4_t*>(p + 8);
  f4_t d = *reinterpret_cast<const f4_t*>(p + 12);
  v16h o;
#pragma unroll
  for (int i = 0; i < 4; ++i) {
    o[i]      = (_Float16)a[i];
    o[4 + i]  = (_Float16)b[i];
    o[8 + i]  = (_Float16)c[i];
    o[12 + i] = (_Float16)d[i];
  }
  return o;
}

// B fragment (32x16) from a [K][N] row-major LDS matrix (strided column reads, for V).
__device__ __forceinline__ v16h load_b_km(const _Float16* base, int stride, int k0, int n0,
                                          int r, int hi) {
  const _Float16* p = base + (k0 + hi * 16) * stride + n0 + r;
  v16h o;
#pragma unroll
  for (int j = 0; j < 16; ++j) o[j] = p[j * stride];
  return o;
}

// C/D layout: c[i] = D[m0 + i + 8*hi][n0 + r]
__device__ __forceinline__ void store_c_f16(_Float16* base, int stride, int m0, int n0,
                                            int r, int hi, v8f c) {
#pragma unroll
  for (int i = 0; i < 8; ++i)
    base[(m0 + i + hi * 8) * stride + n0 + r] = (_Float16)c[i];
}

#define WMMA_F16(a, b, c) \
  __builtin_amdgcn_wmma_f32_16x16x32_f16(false, (a), false, (b), (short)0, (c), false, false)

__global__ __launch_bounds__(256) void leaf_block_attn(
    const float* __restrict__ x, const float* __restrict__ edge,
    const float* __restrict__ qkv_w, const float* __restrict__ qkv_b,
    const float* __restrict__ proj_w, const float* __restrict__ proj_b,
    const float* __restrict__ gate_w, const float* __restrict__ gate_b,
    const float* __restrict__ mask, float* __restrict__ out) {
  __shared__ _Float16 lds[LDS_HALFS];
  const int nb = blockIdx.x, b = blockIdx.y;
  const int tid = threadIdx.x;
  const int wave = tid >> 5;
  const int lane = tid & 31;
  const int r = lane & 15, hi = lane >> 4;

  _Float16* qkvS = lds + OFF_QKV;
  _Float16* xb   = lds + OFF_XO;  // phases 0/1
  _Float16* outS = lds + OFF_XO;  // phases 2/3/4 (aliased; barrier-separated)

  const size_t xbase = ((size_t)b * (kNB * kLQ) + (size_t)nb * kLQ) * kC;

  // ---------- Phase 0: load x block -> f16 LDS ----------
#pragma unroll
  for (int it = 0; it < 16; ++it) {
    int flat4 = tid + it * 256;  // 4096 float4 total
    int row = flat4 >> 6, col4 = flat4 & 63;
    f4_t v = *reinterpret_cast<const f4_t*>(x + xbase + row * kC + col4 * 4);
    h4_t hv;
#pragma unroll
    for (int j = 0; j < 4; ++j) hv[j] = (_Float16)v[j];
    *reinterpret_cast<h4_t*>(xb + row * XS + col4 * 4) = hv;
  }
  __syncthreads();

  // ---------- Phase 1: qkv = xb @ qkv_w^T + qkv_b  (M=64 N=768 K=256) ----------
  for (int j = 0; j < 6; ++j) {
    const int n0 = (wave * 6 + j) * 16;
    v16h bw[8];
#pragma unroll
    for (int ks = 0; ks < 8; ++ks) bw[ks] = load_bt_w(qkv_w, kC, n0, ks * 32, r, hi);
    const float bias = qkv_b[n0 + r];
#pragma unroll
    for (int mt = 0; mt < 4; ++mt) {
      v8f acc = zero8();
#pragma unroll
      for (int ks = 0; ks < 8; ++ks) {
        v16h a = load_a(xb, XS, mt * 16, ks * 32, r, hi);
        acc = WMMA_F16(a, bw[ks], acc);
      }
#pragma unroll
      for (int i = 0; i < 8; ++i) acc[i] += bias;
      store_c_f16(qkvS, QS, mt * 16, n0, r, hi, acc);
    }
  }
  __syncthreads();

  // ---------- Phases 2+3: per-head attention (head = wave) ----------
  const int h = wave;
  const int ccQ = h * kHD, ccK = kC + h * kHD, ccV = 2 * kC + h * kHD;
  const float gw0 = gate_w[h * 4 + 0], gw1 = gate_w[h * 4 + 1];
  const float gw2 = gate_w[h * 4 + 2], gw3 = gate_w[h * 4 + 3];
  const float gb  = gate_b[h];

  v16h kt[4];  // K^T fragments: B[d][kk] = K[kk][d] -> Bt loader on token-major rows
#pragma unroll
  for (int nt = 0; nt < 4; ++nt) kt[nt] = load_bt_lds(qkvS, QS, nt * 16, ccK, r, hi);
  v16h bv[2][2];  // V fragments: [K=token][N=d], strided
#pragma unroll
  for (int nt = 0; nt < 2; ++nt)
#pragma unroll
    for (int ks = 0; ks < 2; ++ks)
      bv[nt][ks] = load_b_km(qkvS, QS, ks * 32, ccV + nt * 16, r, hi);

  _Float16* pW = lds + OFF_P + wave * (16 * PS);
  const float* maskB = mask + (((size_t)b * kNB + nb) * kLQ) * kLQ;
  const float* edgeB = edge + ((size_t)nb * kLQ) * (size_t)kLQ * 4;

  for (int mt = 0; mt < 4; ++mt) {
    const int m0 = mt * 16;
    v16h aq = load_a(qkvS, QS, m0, ccQ, r, hi);
    v8f sS[4];
#pragma unroll
    for (int nt = 0; nt < 4; ++nt) {
      sS[nt] = zero8();
      sS[nt] = WMMA_F16(aq, kt[nt], sS[nt]);
    }

#pragma unroll
    for (int i = 0; i < 8; ++i) {
      const int li = i + hi * 8;  // local row in this 16-row tile
      const int q  = m0 + li;
      float vals[4], gts[4];
#pragma unroll
      for (int nt = 0; nt < 4; ++nt) {
        const int k = nt * 16 + r;
        f4_t e = *reinterpret_cast<const f4_t*>(edgeB + ((size_t)q * kLQ + k) * 4);
        if (k == q && q < 63) { e[0] = 0.f; e[1] = 0.f; e[2] = 0.f; e[3] = 1.f; }
        const float mv = maskB[(size_t)q * kLQ + k];
        const bool on = (mv != 0.f);
        const float sc = sS[nt][i] * kScale + e[3];
        vals[nt] = on ? sc : -__builtin_inff();
        gts[nt]  = on ? (e[0] * gw0 + e[1] * gw1 + e[2] * gw2 + e[3] * gw3 + gb) : 0.f;
      }
      // softmax over the 64-wide row: 4 regs/lane x 16 lanes within this hi-half
      float mx = fmaxf(fmaxf(vals[0], vals[1]), fmaxf(vals[2], vals[3]));
#pragma unroll
      for (int d = 1; d < 16; d <<= 1) mx = fmaxf(mx, __shfl_xor(mx, d, 32));
      float p0 = __expf(vals[0] - mx), p1 = __expf(vals[1] - mx);
      float p2 = __expf(vals[2] - mx), p3 = __expf(vals[3] - mx);
      float se = p0 + p1 + p2 + p3;
#pragma unroll
      for (int d = 1; d < 16; d <<= 1) se += __shfl_xor(se, d, 32);
      const float inv = 1.f / se;
      pW[li * PS +  0 + r] = (_Float16)(p0 * inv + gts[0]);
      pW[li * PS + 16 + r] = (_Float16)(p1 * inv + gts[1]);
      pW[li * PS + 32 + r] = (_Float16)(p2 * inv + gts[2]);
      pW[li * PS + 48 + r] = (_Float16)(p3 * inv + gts[3]);
    }

    // out_h[m0:m0+16, :] = (probs+gate)[m0:m0+16, :] @ V_h   (same-wave LDS RAW)
    v16h a0 = load_a(pW, PS, 0, 0, r, hi);
    v16h a1 = load_a(pW, PS, 0, 32, r, hi);
#pragma unroll
    for (int nt = 0; nt < 2; ++nt) {
      v8f acc = zero8();
      acc = WMMA_F16(a0, bv[nt][0], acc);
      acc = WMMA_F16(a1, bv[nt][1], acc);
      store_c_f16(outS, XS, m0, h * kHD + nt * 16, r, hi, acc);
    }
  }
  __syncthreads();

  // ---------- Phase 4: y = att_out @ proj_w^T + proj_b  (M=64 N=256 K=256) ----------
  for (int j = 0; j < 2; ++j) {
    const int n0 = (wave * 2 + j) * 16;
    v16h bw[8];
#pragma unroll
    for (int ks = 0; ks < 8; ++ks) bw[ks] = load_bt_w(proj_w, kC, n0, ks * 32, r, hi);
    const float bias = proj_b[n0 + r];
#pragma unroll
    for (int mt = 0; mt < 4; ++mt) {
      v8f acc = zero8();
#pragma unroll
      for (int ks = 0; ks < 8; ++ks) {
        v16h a = load_a(outS, XS, mt * 16, ks * 32, r, hi);
        acc = WMMA_F16(a, bw[ks], acc);
      }
#pragma unroll
      for (int i = 0; i < 8; ++i)
        out[xbase + (size_t)(mt * 16 + i + hi * 8) * kC + n0 + r] = acc[i] + bias;
    }
  }
}

}  // namespace

extern "C" void kernel_launch(void* const* d_in, const int* in_sizes, int n_in,
                              void* d_out, int out_size, void* d_ws, size_t ws_size,
                              hipStream_t stream) {
  (void)n_in; (void)out_size; (void)d_ws; (void)ws_size;
  const float* x      = (const float*)d_in[0];
  const float* edge   = (const float*)d_in[1];
  const float* qkv_w  = (const float*)d_in[2];
  const float* qkv_b  = (const float*)d_in[3];
  const float* proj_w = (const float*)d_in[4];
  const float* proj_b = (const float*)d_in[5];
  const float* gate_w = (const float*)d_in[6];
  const float* gate_b = (const float*)d_in[7];
  const float* mask   = (const float*)d_in[8];
  float* out = (float*)d_out;

  const int B = in_sizes[0] / (kNB * kLQ * kC);  // = 4
  dim3 grid(kNB, B);
  leaf_block_attn<<<grid, 256, 0, stream>>>(x, edge, qkv_w, qkv_b, proj_w, proj_b,
                                            gate_w, gate_b, mask, out);
}
